// GCNModel_87205015978674
// MI455X (gfx1250) — compile-verified, matrix operand order checked by reading, not run on hardware
//
#include <hip/hip_runtime.h>
#include <hip/hip_bf16.h>

// ---------------------------------------------------------------------------
// GCN forward for MI455X (gfx1250): bf16 WMMA GEMMs + L2-resident edge scatter
// ---------------------------------------------------------------------------

#define N_NODES 50000
#define N_EDGES 600000
#define IN_DIM  64
#define H_DIM   256
#define N_LAYERS 3
#define OUT_DIM 12
#define N_GRAPHS 64
#define BN_EPS 1e-5f

typedef __attribute__((ext_vector_type(16))) __bf16 v16bf;
typedef __attribute__((ext_vector_type(8)))  __bf16 v8bf;
typedef __attribute__((ext_vector_type(8)))  float  v8f;

// float -> bf16 bits (round to nearest even)
__device__ __forceinline__ unsigned short f2bf(float f) {
    union { float f; unsigned u; } v; v.f = f;
    unsigned u = v.u;
    unsigned r = u + 0x7FFFu + ((u >> 16) & 1u);
    return (unsigned short)(r >> 16);
}

// Load a 16x32 bf16 A/B fragment slice for this lane: two contiguous 16B loads.
// Lane layout per CDNA5 ISA 7.12.2: lanes 0-15 hold K = {0..7, 16..23},
// lanes 16-31 hold K = {8..15, 24..31} (kb pre-applied by caller).
__device__ __forceinline__ v16bf load_frag(const unsigned short* p) {
    v8bf lo = *(const v8bf*)(p);
    v8bf hi = *(const v8bf*)(p + 16);
    v16bf r;
#pragma unroll
    for (int i = 0; i < 8; ++i) { r[i] = lo[i]; r[i + 8] = hi[i]; }
    return r;
}

// ---------------------------------------------------------------------------
// Degree / norm precompute
// ---------------------------------------------------------------------------
__global__ void k_deg_init(float* deg) {
    int i = blockIdx.x * blockDim.x + threadIdx.x;
    if (i < N_NODES) deg[i] = 1.0f;               // self loop
}
__global__ void k_deg_acc(const int* __restrict__ ei, float* deg) {
    int e = blockIdx.x * blockDim.x + threadIdx.x;
    if (e < N_EDGES) atomicAdd(&deg[ei[N_EDGES + e]], 1.0f);   // col = target
}
__global__ void k_rsqrt(float* deg) {
    int i = blockIdx.x * blockDim.x + threadIdx.x;
    if (i < N_NODES) deg[i] = rsqrtf(deg[i]);
}
__global__ void k_norm(const int* __restrict__ ei, const float* __restrict__ dis,
                       float* __restrict__ norm) {
    int e = blockIdx.x * blockDim.x + threadIdx.x;
    if (e < N_EDGES) norm[e] = dis[ei[e]] * dis[ei[N_EDGES + e]];
}

// ---------------------------------------------------------------------------
// Conversions
// ---------------------------------------------------------------------------
__global__ void k_cvt_bf16(const float* __restrict__ src, unsigned short* __restrict__ dst,
                           int n) {
    int i = blockIdx.x * blockDim.x + threadIdx.x;
    if (i < n) dst[i] = f2bf(src[i]);
}
// W [K, Hout] fp32 row-major  ->  Wt [Hout, K] bf16 (B fragments become contiguous)
__global__ void k_cvt_wt(const float* __restrict__ w, unsigned short* __restrict__ wt,
                         int K, int Hout) {
    int i = blockIdx.x * blockDim.x + threadIdx.x;
    if (i < K * Hout) {
        int n = i / K, k = i - n * K;
        wt[n * K + k] = f2bf(w[k * Hout + n]);
    }
}

// ---------------------------------------------------------------------------
// WMMA GEMM: C[M,256] = A[M,K](bf16) * Bt[256,K](bf16)^T (+bias)
// Block = 256 threads = 8 waves; wave tile = 16 rows x 64 cols; block = 32x256.
// ---------------------------------------------------------------------------
__global__ __launch_bounds__(256) void k_gemm_bf16(
    const unsigned short* __restrict__ A,
    const unsigned short* __restrict__ Bt,
    const float* __restrict__ bias,          // nullable
    float* __restrict__ C,
    unsigned short* __restrict__ Cbf,        // nullable: also emit bf16 copy
    int M, int K) {
    const int lane = threadIdx.x & 31;
    const int wave = threadIdx.x >> 5;
    const int mtile = blockIdx.x * 2 + (wave >> 2);
    if (mtile * 16 >= M) return;             // wave-uniform guard (EXEC stays all-1)
    const int n0 = (wave & 3) * 64;
    const int rowA = mtile * 16 + (lane & 15);
    const int kb = (lane >> 4) * 8;          // K-half select per ISA layout

    v8f acc0 = (v8f)0.0f, acc1 = (v8f)0.0f, acc2 = (v8f)0.0f, acc3 = (v8f)0.0f;

    for (int k0 = 0; k0 < K; k0 += 32) {
        v16bf a = load_frag(A + (size_t)rowA * K + k0 + kb);
        v16bf b0 = load_frag(Bt + (size_t)(n0 +  0 + (lane & 15)) * K + k0 + kb);
        v16bf b1 = load_frag(Bt + (size_t)(n0 + 16 + (lane & 15)) * K + k0 + kb);
        v16bf b2 = load_frag(Bt + (size_t)(n0 + 32 + (lane & 15)) * K + k0 + kb);
        v16bf b3 = load_frag(Bt + (size_t)(n0 + 48 + (lane & 15)) * K + k0 + kb);
        acc0 = __builtin_amdgcn_wmma_f32_16x16x32_bf16(false, a, false, b0, (short)0, acc0, false, false);
        acc1 = __builtin_amdgcn_wmma_f32_16x16x32_bf16(false, a, false, b1, (short)0, acc1, false, false);
        acc2 = __builtin_amdgcn_wmma_f32_16x16x32_bf16(false, a, false, b2, (short)0, acc2, false, false);
        acc3 = __builtin_amdgcn_wmma_f32_16x16x32_bf16(false, a, false, b3, (short)0, acc3, false, false);
    }

    // C/D layout: VGPR i -> row i (lanes 0-15) / row 8+i (lanes 16-31), col = lane&15
    const int rbase = mtile * 16 + (lane >> 4) * 8;
    v8f accs[4] = {acc0, acc1, acc2, acc3};
#pragma unroll
    for (int j = 0; j < 4; ++j) {
        const int c = n0 + j * 16 + (lane & 15);
        const float bb = bias ? bias[c] : 0.0f;
#pragma unroll
        for (int i = 0; i < 8; ++i) {
            float val = accs[j][i] + bb;
            size_t idx = (size_t)(rbase + i) * H_DIM + c;
            C[idx] = val;
            if (Cbf) Cbf[idx] = f2bf(val);
        }
    }
}

// ---------------------------------------------------------------------------
// agg init: self-loop contribution + conv bias (folds the N self-loop edges)
// ---------------------------------------------------------------------------
__global__ void k_agg_init(const float* __restrict__ hw, const float* __restrict__ dis,
                           const float* __restrict__ cbias, float* __restrict__ agg) {
    int idx = blockIdx.x * blockDim.x + threadIdx.x;
    if (idx < N_NODES * H_DIM) {
        int i = idx >> 8, f = idx & 255;
        float s = dis[i];
        agg[idx] = hw[idx] * s * s + cbias[f];
    }
}

// ---------------------------------------------------------------------------
// Edge scatter: agg[col] += hw[row] * norm[e].  32 threads / edge, 8 feats each.
// hw and agg are L2-resident (51 MB each, 192 MB L2) -> atomics stay on-chip.
// ---------------------------------------------------------------------------
__global__ void k_scatter(const int* __restrict__ ei, const float* __restrict__ norm,
                          const float* __restrict__ hw, float* __restrict__ agg) {
    int idx = blockIdx.x * blockDim.x + threadIdx.x;
    int e = idx >> 5;
    if (e >= N_EDGES) return;
    int f0 = (idx & 31) * 8;
    int r = ei[e], c = ei[N_EDGES + e];
    float w = norm[e];
    const float4* src = (const float4*)(hw + (size_t)r * H_DIM + f0);
    float4 a = src[0], b = src[1];
    float* dst = agg + (size_t)c * H_DIM + f0;
    atomicAdd(dst + 0, a.x * w); atomicAdd(dst + 1, a.y * w);
    atomicAdd(dst + 2, a.z * w); atomicAdd(dst + 3, a.w * w);
    atomicAdd(dst + 4, b.x * w); atomicAdd(dst + 5, b.y * w);
    atomicAdd(dst + 6, b.z * w); atomicAdd(dst + 7, b.w * w);
}

// ---------------------------------------------------------------------------
// BatchNorm (training stats) + ReLU; also emits bf16 h for the next WMMA GEMM
// ---------------------------------------------------------------------------
__global__ void k_stats_zero(float* stats) {
    int i = blockIdx.x * blockDim.x + threadIdx.x;
    if (i < 2 * H_DIM) stats[i] = 0.0f;
}
__global__ __launch_bounds__(256) void k_stats_acc(const float* __restrict__ agg,
                                                   float* __restrict__ stats) {
    const int f = threadIdx.x;                 // one feature per thread: coalesced rows
    const int r0 = blockIdx.x * 64;
    float s = 0.0f, s2 = 0.0f;
#pragma unroll 4
    for (int r = 0; r < 64; ++r) {
        int row = r0 + r;
        if (row < N_NODES) {
            float v = agg[(size_t)row * H_DIM + f];
            s += v; s2 += v * v;
        }
    }
    atomicAdd(&stats[f], s);
    atomicAdd(&stats[H_DIM + f], s2);
}
__global__ void k_bn_relu(const float* __restrict__ agg, const float* __restrict__ stats,
                          const float* __restrict__ g, const float* __restrict__ b,
                          unsigned short* __restrict__ h_bf, float* __restrict__ h_f32) {
    int idx = blockIdx.x * blockDim.x + threadIdx.x;
    if (idx < N_NODES * H_DIM) {
        int f = idx & 255;
        const float invN = 1.0f / (float)N_NODES;
        float mean = stats[f] * invN;
        float var = stats[H_DIM + f] * invN - mean * mean;
        float y = (agg[idx] - mean) * rsqrtf(var + BN_EPS) * g[f] + b[f];
        y = fmaxf(y, 0.0f);
        h_bf[idx] = f2bf(y);
        h_f32[idx] = y;
    }
}

// ---------------------------------------------------------------------------
// Global add pool + MLP head
// ---------------------------------------------------------------------------
__global__ void k_zero(float* p, int n) {
    int i = blockIdx.x * blockDim.x + threadIdx.x;
    if (i < n) p[i] = 0.0f;
}
__global__ void k_pool(const int* __restrict__ batch, const float* __restrict__ h,
                       float* __restrict__ g) {
    int idx = blockIdx.x * blockDim.x + threadIdx.x;
    if (idx >= N_NODES * (H_DIM / 4)) return;
    int node = idx >> 6;
    int f0 = (idx & 63) * 4;
    int bg = batch[node];
    float4 v = *(const float4*)(h + (size_t)node * H_DIM + f0);
    float* dst = g + (size_t)bg * H_DIM + f0;
    atomicAdd(dst + 0, v.x); atomicAdd(dst + 1, v.y);
    atomicAdd(dst + 2, v.z); atomicAdd(dst + 3, v.w);
}
__global__ __launch_bounds__(128) void k_head(const float* __restrict__ g,
    const float* __restrict__ w1, const float* __restrict__ b1,
    const float* __restrict__ w2, const float* __restrict__ b2,
    float* __restrict__ out) {
    __shared__ float z[H_DIM / 2];
    const int gi = blockIdx.x, t = threadIdx.x;
    float acc = b1[t];
    for (int k = 0; k < H_DIM; ++k)
        acc += g[(size_t)gi * H_DIM + k] * w1[k * (H_DIM / 2) + t];
    z[t] = fmaxf(acc, 0.0f);
    __syncthreads();
    if (t < OUT_DIM) {
        float a2 = b2[t];
        for (int k = 0; k < H_DIM / 2; ++k)
            a2 += z[k] * w2[k * OUT_DIM + t];
        out[gi * OUT_DIM + t] = a2;
    }
}

// ---------------------------------------------------------------------------
// Host launch
// ---------------------------------------------------------------------------
extern "C" void kernel_launch(void* const* d_in, const int* in_sizes, int n_in,
                              void* d_out, int out_size, void* d_ws, size_t ws_size,
                              hipStream_t stream) {
    (void)in_sizes; (void)n_in; (void)out_size; (void)ws_size;
    const float* x       = (const float*)d_in[0];
    const int*   ei      = (const int*)d_in[1];
    const int*   batch   = (const int*)d_in[2];
    const float* node_w  = (const float*)d_in[3];
    const float* node_b  = (const float*)d_in[4];
    const float* conv_w  = (const float*)d_in[5];
    const float* conv_b  = (const float*)d_in[6];
    const float* bn_g    = (const float*)d_in[7];
    const float* bn_b    = (const float*)d_in[8];
    const float* head_w1 = (const float*)d_in[9];
    const float* head_b1 = (const float*)d_in[10];
    const float* head_w2 = (const float*)d_in[11];
    const float* head_b2 = (const float*)d_in[12];
    float* out = (float*)d_out;

    // workspace carve-up (256B aligned)
    char* base = (char*)d_ws; size_t off = 0;
    auto carve = [&](size_t bytes) { void* p = base + off; off += (bytes + 255) & ~(size_t)255; return p; };
    float*          dis   = (float*)carve((size_t)N_NODES * 4);
    float*          norm  = (float*)carve((size_t)N_EDGES * 4);
    unsigned short* xb    = (unsigned short*)carve((size_t)N_NODES * IN_DIM * 2);
    unsigned short* hA    = (unsigned short*)carve((size_t)N_NODES * H_DIM * 2);
    unsigned short* wt    = (unsigned short*)carve((size_t)H_DIM * H_DIM * 2);
    float*          hw    = (float*)carve((size_t)N_NODES * H_DIM * 4);
    float*          agg   = (float*)carve((size_t)N_NODES * H_DIM * 4);
    float*          stats = (float*)carve(2 * H_DIM * 4);
    float*          gsum  = (float*)carve((size_t)N_GRAPHS * H_DIM * 4);

    const int T = 256;
    auto gb = [](long n, int t) { return dim3((unsigned)((n + t - 1) / t)); };

    // --- degree / symmetric norm ---
    k_deg_init<<<gb(N_NODES, T), T, 0, stream>>>(dis);
    k_deg_acc <<<gb(N_EDGES, T), T, 0, stream>>>(ei, dis);
    k_rsqrt   <<<gb(N_NODES, T), T, 0, stream>>>(dis);
    k_norm    <<<gb(N_EDGES, T), T, 0, stream>>>(ei, dis, norm);

    // --- input projection: h = x @ node_w + node_b  (bf16 WMMA) ---
    k_cvt_bf16<<<gb((long)N_NODES * IN_DIM, T), T, 0, stream>>>(x, xb, N_NODES * IN_DIM);
    k_cvt_wt  <<<gb((long)IN_DIM * H_DIM, T), T, 0, stream>>>(node_w, wt, IN_DIM, H_DIM);
    k_gemm_bf16<<<gb((long)N_NODES / 32 + 1, 1), T, 0, stream>>>(
        xb, wt, node_b, hw, hA, N_NODES, IN_DIM);

    // --- GCN layers ---
    for (int l = 0; l < N_LAYERS; ++l) {
        k_cvt_wt<<<gb((long)H_DIM * H_DIM, T), T, 0, stream>>>(
            conv_w + (size_t)l * H_DIM * H_DIM, wt, H_DIM, H_DIM);
        k_gemm_bf16<<<gb((long)N_NODES / 32 + 1, 1), T, 0, stream>>>(
            hA, wt, nullptr, hw, nullptr, N_NODES, H_DIM);
        k_agg_init<<<gb((long)N_NODES * H_DIM, T), T, 0, stream>>>(
            hw, dis, conv_b + (size_t)l * H_DIM, agg);
        k_scatter<<<gb((long)N_EDGES * 32, T), T, 0, stream>>>(ei, norm, hw, agg);
        k_stats_zero<<<gb(2 * H_DIM, T), T, 0, stream>>>(stats);
        k_stats_acc<<<gb((N_NODES + 63) / 64, 1), T, 0, stream>>>(agg, stats);
        k_bn_relu<<<gb((long)N_NODES * H_DIM, T), T, 0, stream>>>(
            agg, stats, bn_g + (size_t)l * H_DIM, bn_b + (size_t)l * H_DIM, hA, hw);
    }

    // --- pool + head ---
    k_zero<<<gb(N_GRAPHS * H_DIM, T), T, 0, stream>>>(gsum, N_GRAPHS * H_DIM);
    k_pool<<<gb((long)N_NODES * (H_DIM / 4), T), T, 0, stream>>>(batch, hw, gsum);
    k_head<<<dim3(N_GRAPHS), 128, 0, stream>>>(gsum, head_w1, head_b1, head_w2, head_b2, out);
}
